// SparseMoE_49804440764916
// MI455X (gfx1250) — matrix operand order repaired and checked by reading, not compiled, not run on hardware
//
#include <hip/hip_runtime.h>

// ---------------- problem constants ----------------
constexpr int B = 16384;
constexpr int D = 512;
constexpr int E = 8;
constexpr int KTOP = 2;

// GEMM tiling
constexpr int BM = 128;   // tokens per tile
constexpr int BN = 128;   // output cols per tile
constexpr int KC = 64;    // k-chunk staged in LDS
constexpr int LDA = 72;   // padded LDS row stride (elems) -> 144B, 16B aligned

typedef __attribute__((ext_vector_type(16))) __bf16 v16bf;
typedef __attribute__((ext_vector_type(8)))  float  v8f;

union FragBF { uint4 q[2]; v16bf v; };

// float -> bf16 (round to nearest even; inputs are normal floats)
__device__ __forceinline__ unsigned short f2bf(float f) {
    unsigned u = __float_as_uint(f);
    unsigned r = u + 0x7FFFu + ((u >> 16) & 1u);
    return (unsigned short)(r >> 16);
}

// low 32 bits of a generic pointer into LDS == allocation-relative LDS byte address
__device__ __forceinline__ unsigned lds_u32(const void* p) {
    return (unsigned)(unsigned long long)p;
}

// async global->LDS 16B copy, GVS addressing: mem = saddr + vaddr + imm ; lds = vdst + imm
#define ASYNC_LD128(ldsoff, voff, sbase, IMM)                                   \
    asm volatile("global_load_async_to_lds_b128 %0, %1, %2 offset:" #IMM        \
                 :: "v"(ldsoff), "v"(voff), "s"(sbase) : "memory")

// ---------------- kernel 1: zero output + cursors ----------------
__global__ void zero_out_kernel(float4* __restrict__ out4, int n4, int* __restrict__ counts) {
    int i = blockIdx.x * blockDim.x + threadIdx.x;
    if (i < n4) out4[i] = make_float4(0.f, 0.f, 0.f, 0.f);
    if (blockIdx.x == 0 && threadIdx.x < E) counts[threadIdx.x] = 0;
}

// ---------------- kernel 2: x f32 -> bf16 ----------------
__global__ void convert_x_kernel(const float4* __restrict__ x4, uint2* __restrict__ xb2, int n4) {
    int i = blockIdx.x * blockDim.x + threadIdx.x;
    if (i >= n4) return;
    float4 v = x4[i];
    uint2 o;
    o.x = (unsigned)f2bf(v.x) | ((unsigned)f2bf(v.y) << 16);
    o.y = (unsigned)f2bf(v.z) | ((unsigned)f2bf(v.w) << 16);
    xb2[i] = o;
}

// ---------------- kernel 3: We[e][k][n] -> WeT bf16 [e][n][k] ----------------
__global__ __launch_bounds__(256) void transpose_we_kernel(const float* __restrict__ We,
                                                           unsigned short* __restrict__ wt) {
    __shared__ float tile[32][33];
    int e  = blockIdx.z;
    int n0 = blockIdx.x * 32;   // n tile
    int k0 = blockIdx.y * 32;   // k tile
    int lx = threadIdx.x & 31;
    int ly = threadIdx.x >> 5;  // 0..7
#pragma unroll
    for (int p = 0; p < 4; ++p) {
        int kl = ly + p * 8;
        tile[kl][lx] = We[((size_t)(e * D) + (k0 + kl)) * D + (n0 + lx)];
    }
    __syncthreads();
#pragma unroll
    for (int p = 0; p < 4; ++p) {
        int nl = ly + p * 8;
        // wt[e][n0+nl][k0+lx] = We[e][k0+lx][n0+nl]
        wt[((size_t)(e * D) + (n0 + nl)) * D + (k0 + lx)] = f2bf(tile[lx][nl]);
    }
}

// ---------------- kernel 4: gating (one wave32 per token) ----------------
__global__ __launch_bounds__(256) void gating_kernel(const float* __restrict__ x,
                                                     const float* __restrict__ Wg,
                                                     const float* __restrict__ bg,
                                                     int*   __restrict__ counts,
                                                     int*   __restrict__ lists,
                                                     float* __restrict__ lw,
                                                     int*   __restrict__ topi) {
    int wv   = threadIdx.x >> 5;
    int lane = threadIdx.x & 31;
    int b    = blockIdx.x * 8 + wv;

    float acc[E];
#pragma unroll
    for (int e = 0; e < E; ++e) acc[e] = 0.f;

    const float* xrow = x + (size_t)b * D;
#pragma unroll 4
    for (int i = 0; i < D / 32; ++i) {
        int d = lane + i * 32;
        float xv = xrow[d];
#pragma unroll
        for (int e = 0; e < E; ++e) acc[e] += xv * Wg[e * D + d];
    }
    // cross-lane reduce within wave32
#pragma unroll
    for (int e = 0; e < E; ++e) {
        float v = acc[e];
        v += __shfl_xor(v, 1);
        v += __shfl_xor(v, 2);
        v += __shfl_xor(v, 4);
        v += __shfl_xor(v, 8);
        v += __shfl_xor(v, 16);
        acc[e] = v + bg[e];
    }
    if (lane == 0) {
        // top-2 with first-index tie-break (matches lax.top_k)
        int i0 = 0; float v0 = acc[0];
#pragma unroll
        for (int e = 1; e < E; ++e) if (acc[e] > v0) { v0 = acc[e]; i0 = e; }
        int i1 = -1; float v1 = -3.4e38f;
#pragma unroll
        for (int e = 0; e < E; ++e) if (e != i0 && acc[e] > v1) { v1 = acc[e]; i1 = e; }
        float t  = __expf(v1 - v0);        // v1 <= v0, stable
        float w0 = 1.f / (1.f + t);
        float w1 = t / (1.f + t);
        topi[b * KTOP + 0] = i0;
        topi[b * KTOP + 1] = i1;
        int p0 = atomicAdd(&counts[i0], 1);
        lists[i0 * B + p0] = b; lw[i0 * B + p0] = w0;
        int p1 = atomicAdd(&counts[i1], 1);
        lists[i1 * B + p1] = b; lw[i1 * B + p1] = w1;
    }
}

// ---------------- kernel 5: gathered expert GEMM, WMMA bf16 + async LDS pipeline ----------------
__global__ __launch_bounds__(256) void moe_gemm_kernel(const unsigned short* __restrict__ xb,
                                                       const unsigned short* __restrict__ wt,
                                                       const float* __restrict__ be,
                                                       const int*   __restrict__ lists,
                                                       const float* __restrict__ lw,
                                                       const int*   __restrict__ counts,
                                                       float* __restrict__ out) {
    __shared__ unsigned short sa[2][BM * LDA];   // gathered x tile [m][k], double buffered
    __shared__ unsigned short sb[2][BN * LDA];   // WeT tile       [n][k], double buffered

    const int e  = blockIdx.z;
    const int n0 = blockIdx.y * BN;
    const int t  = blockIdx.x;
    const int cnt = counts[e];
    if (t * BM >= cnt) return;   // uniform across block

    const int tid  = threadIdx.x;
    const int lane = tid & 31;
    const int w    = tid >> 5;          // wave id 0..7 -> owns M rows [16w,16w+16)

    // staging mapping: two threads per row, each moves 64 contiguous bytes per tile
    const int sr = tid >> 1;            // row 0..127
    const int sh = tid & 1;             // half 0/1 -> k elem offset 0/32
    const int mrow_s = t * BM + sr;
    const int btok = (mrow_s < cnt) ? lists[e * B + mrow_s] : 0;

    const unsigned long long xb64 = (unsigned long long)xb;
    const unsigned long long wt64 = (unsigned long long)wt;
    // global byte offsets at k0 = 0
    unsigned voffA = (unsigned)(((size_t)btok * D + sh * 32) * 2);
    unsigned voffB = (unsigned)((((size_t)(e * D + (n0 + sr))) * D + sh * 32) * 2);
    const unsigned kstepB = KC * 2;     // byte advance per k-chunk
    // LDS byte offsets (same for both buffers modulo buffer base)
    unsigned ldsA[2], ldsB[2];
    ldsA[0] = lds_u32(&sa[0][sr * LDA + sh * 32]);
    ldsA[1] = lds_u32(&sa[1][sr * LDA + sh * 32]);
    ldsB[0] = lds_u32(&sb[0][sr * LDA + sh * 32]);
    ldsB[1] = lds_u32(&sb[1][sr * LDA + sh * 32]);

    v8f acc[8];
#pragma unroll
    for (int nf = 0; nf < 8; ++nf)
#pragma unroll
        for (int i = 0; i < 8; ++i) acc[nf][i] = 0.f;

    // ---- prologue: async-issue chunk 0 into buffer 0 (8 x b128 per thread) ----
    ASYNC_LD128(ldsA[0], voffA, xb64, 0);
    ASYNC_LD128(ldsA[0], voffA, xb64, 16);
    ASYNC_LD128(ldsA[0], voffA, xb64, 32);
    ASYNC_LD128(ldsA[0], voffA, xb64, 48);
    ASYNC_LD128(ldsB[0], voffB, wt64, 0);
    ASYNC_LD128(ldsB[0], voffB, wt64, 16);
    ASYNC_LD128(ldsB[0], voffB, wt64, 32);
    ASYNC_LD128(ldsB[0], voffB, wt64, 48);

    for (int k0 = 0; k0 < D; k0 += KC) {
        const int buf = (k0 / KC) & 1;
        if (k0 + KC < D) {
            // issue next chunk into the other buffer (safe: barrier at end of prev iter)
            const int nb = buf ^ 1;
            unsigned nA = voffA + kstepB * (unsigned)(k0 / KC + 1);
            unsigned nB = voffB + kstepB * (unsigned)(k0 / KC + 1);
            ASYNC_LD128(ldsA[nb], nA, xb64, 0);
            ASYNC_LD128(ldsA[nb], nA, xb64, 16);
            ASYNC_LD128(ldsA[nb], nA, xb64, 32);
            ASYNC_LD128(ldsA[nb], nA, xb64, 48);
            ASYNC_LD128(ldsB[nb], nB, wt64, 0);
            ASYNC_LD128(ldsB[nb], nB, wt64, 16);
            ASYNC_LD128(ldsB[nb], nB, wt64, 32);
            ASYNC_LD128(ldsB[nb], nB, wt64, 48);
            // previous chunk (8 older async ops) must have landed
            asm volatile("s_wait_asynccnt 0x8" ::: "memory");
        } else {
            asm volatile("s_wait_asynccnt 0x0" ::: "memory");
        }
        __syncthreads();

        const unsigned short* cur_sa = sa[buf];
        const unsigned short* cur_sb = sb[buf];
#pragma unroll
        for (int kk = 0; kk < KC; kk += 32) {
            // A fragment (16x32 bf16): lane<16 holds K {0..7,16..23}, lane>=16 {8..15,24..31}
            const int arow  = w * 16 + (lane & 15);
            const int abase = kk + ((lane < 16) ? 0 : 8);
            FragBF a;
            a.q[0] = *(const uint4*)(cur_sa + arow * LDA + abase);
            a.q[1] = *(const uint4*)(cur_sa + arow * LDA + abase + 16);
            // batch-load all 8 B fragments, then issue 8 WMMAs back-to-back
            const int bbase = kk + ((lane < 16) ? 0 : 8);
            FragBF bfr[8];
#pragma unroll
            for (int nf = 0; nf < 8; ++nf) {
                const int bcol = nf * 16 + (lane & 15);
                bfr[nf].q[0] = *(const uint4*)(cur_sb + bcol * LDA + bbase);
                bfr[nf].q[1] = *(const uint4*)(cur_sb + bcol * LDA + bbase + 16);
            }
#pragma unroll
            for (int nf = 0; nf < 8; ++nf) {
                acc[nf] = __builtin_amdgcn_wmma_f32_16x16x32_bf16(
                    false, a.v, false, bfr[nf].v, (short)0, acc[nf], false, false);
            }
        }
        __syncthreads();
    }

    // ---- epilogue: C layout lane l, vgpr r -> M = r + 8*(l>>4), N = l&15 ----
    const int mbase = t * BM + w * 16 + ((lane >> 4) << 3);
    int   tok[8];
    float wv8[8];
#pragma unroll
    for (int r = 0; r < 8; ++r) {
        int mrow = mbase + r;
        bool valid = mrow < cnt;
        tok[r] = valid ? lists[e * B + mrow] : -1;
        wv8[r] = valid ? lw[e * B + mrow] : 0.f;
    }
#pragma unroll
    for (int nf = 0; nf < 8; ++nf) {
        const int col = n0 + nf * 16 + (lane & 15);
        const float bev = be[e * D + col];
#pragma unroll
        for (int r = 0; r < 8; ++r) {
            if (tok[r] >= 0) {
                atomicAdd(out + (size_t)tok[r] * D + col, (acc[nf][r] + bev) * wv8[r]);
            }
        }
    }
}

// ---------------- launcher ----------------
extern "C" void kernel_launch(void* const* d_in, const int* in_sizes, int n_in,
                              void* d_out, int out_size, void* d_ws, size_t ws_size,
                              hipStream_t stream) {
    const float* x  = (const float*)d_in[0];   // (B, D)
    const float* Wg = (const float*)d_in[1];   // (E, D)
    const float* bg = (const float*)d_in[2];   // (E,)
    const float* We = (const float*)d_in[3];   // (E, D, D)
    const float* be = (const float*)d_in[4];   // (E, D)

    float* out  = (float*)d_out;                        // (B, D) f32
    int*   topi = (int*)(out + (size_t)B * D);          // (B, K) i32, concatenated

    // workspace layout
    char* ws = (char*)d_ws;
    unsigned short* xb = (unsigned short*)ws;                          // B*D bf16   = 16 MiB
    unsigned short* wt = (unsigned short*)(ws + (size_t)B * D * 2);    // E*D*D bf16 =  4 MiB
    char* p2 = ws + (size_t)B * D * 2 + (size_t)E * D * D * 2;
    int*   lists  = (int*)p2;                                          // E*B ints   = 512 KiB
    float* lwgt   = (float*)(p2 + (size_t)E * B * 4);                  // E*B floats = 512 KiB
    int*   counts = (int*)(p2 + (size_t)E * B * 8);                    // E ints

    const int n4 = (B * D) / 4;

    zero_out_kernel<<<(n4 + 255) / 256, 256, 0, stream>>>((float4*)out, n4, counts);
    convert_x_kernel<<<(n4 + 255) / 256, 256, 0, stream>>>((const float4*)x, (uint2*)xb, n4);
    transpose_we_kernel<<<dim3(D / 32, D / 32, E), 256, 0, stream>>>(We, wt);
    gating_kernel<<<B / 8, 256, 0, stream>>>(x, Wg, bg, counts, lists, lwgt, topi);
    moe_gemm_kernel<<<dim3(B / BM, D / BN, E), 256, 0, stream>>>(xb, wt, be, lists, lwgt, counts, out);
}
